// GIN_27908697489545
// MI455X (gfx1250) — matrix operand order repaired and checked by reading, not compiled, not run on hardware
//
#include <hip/hip_runtime.h>
#include <hip/hip_bf16.h>

// ---------------------------------------------------------------------------
// GIN (3-layer) for MI455X / gfx1250, wave32.
// Roofline: edge scatter/gather dominates (~1.4 GB traffic, ~205M f32 atomics;
// ~60us at 23.3 TB/s; whole ~100MB working set is L2-resident on 192MB L2).
// Dense MLPs use v_wmma_f32_16x16x32_bf16 with f32 accumulation; weights are
// pre-packed to transposed bf16 so B fragments are single 32B contiguous loads.
// ---------------------------------------------------------------------------

typedef __attribute__((ext_vector_type(16))) __bf16 v16bf;
typedef __attribute__((ext_vector_type(8)))  float  v8f;

// ---------------------------------------------------------------------------
// pack_w: wT[n*K + k] = bf16( w[k*32 + n] )   (w is [K x 32] f32, row-major)
// ---------------------------------------------------------------------------
__global__ void pack_w(const float* __restrict__ w, __bf16* __restrict__ wT, int K) {
    int i = blockIdx.x * blockDim.x + threadIdx.x;      // i = n*K + k
    if (i >= K * 32) return;
    int n = i / K, k = i - n * K;
    wT[i] = (__bf16)w[k * 32 + n];
}

// ---------------------------------------------------------------------------
// scatter_add: agg[dst, :] += in[src, :]   (one wave per edge, lane = channel)
// unsafeAtomicAdd -> global_atomic_add_f32 (L2-resident lines).
// ---------------------------------------------------------------------------
__global__ void scatter_add(const float* __restrict__ in,
                            float* __restrict__ agg,
                            const int* __restrict__ edges,
                            int nEdges, int C) {
    long long gid = (long long)blockIdx.x * blockDim.x + threadIdx.x;
    int e    = (int)(gid >> 5);
    int lane = (int)(gid & 31);
    if (e >= nEdges) return;
    int src = edges[e];            // edge_index row 0
    int dst = edges[nEdges + e];   // edge_index row 1
    const float* srow = in  + (size_t)src * C;
    float*       drow = agg + (size_t)dst * C;
    for (int c = lane; c < C; c += 32) {
        unsafeAtomicAdd(&drow[c], srow[c]);
    }
}

// ---------------------------------------------------------------------------
// One wave computes a 16-row tile of: out = relu(in @ A + bA) @ B + bB
//   in  : [M x K1] f32
//   wAT : [32 x K1] bf16 (transposed, packed)   wBT : [32 x 32] bf16
// A fragment: lane holds row M=lane%16, K halves split at lane 16 (element j holds
// k = 16*hi + j; B uses the same order). B fragment (from wT): 16 contiguous bf16 at [n][kbase] -> v16bf.
// C/D: VGPR r -> row r (lanes 0-15) / r+8 (lanes 16-31), col = lane%16.
// ---------------------------------------------------------------------------
template<int K1>
__global__ void mlp_kernel(const float* __restrict__ in,
                           const __bf16* __restrict__ wAT, const float* __restrict__ bA,
                           const __bf16* __restrict__ wBT, const float* __restrict__ bB,
                           float* __restrict__ out, int M) {
    const int lane = threadIdx.x;       // 0..31
    const int hi   = lane >> 4;
    const int mloc = lane & 15;
    const int tile = blockIdx.x;

    int rowIdx = tile * 16 + mloc;
    if (rowIdx >= M) rowIdx = M - 1;    // clamp loads (stores guarded)
    const float* arow = in + (size_t)rowIdx * K1;

    __shared__ float sT[16][33];        // intermediate 16x32, +1 pad

    float ba0 = bA[mloc], ba1 = bA[16 + mloc];
    v8f acc0, acc1;
#pragma unroll
    for (int i = 0; i < 8; ++i) { acc0[i] = ba0; acc1[i] = ba1; }

    // ---- GEMM 1: K1 in steps of 32 ----
#pragma unroll
    for (int k0 = 0; k0 < K1; k0 += 32) {
        // A fragment (two contiguous 32B f32 segments -> bf16)
        v16bf a;
        const int ka = k0 + hi * 16;
#pragma unroll
        for (int j = 0; j < 8; ++j) {
            a[j]     = (__bf16)arow[ka + j];
            a[j + 8] = (__bf16)arow[ka + 8 + j];
        }
        // B fragments: single 32B contiguous bf16 loads from packed weights
        const int kb = k0 + hi * 16;
        v16bf b0 = *reinterpret_cast<const v16bf*>(wAT + (size_t)mloc * K1 + kb);
        v16bf b1 = *reinterpret_cast<const v16bf*>(wAT + (size_t)(mloc + 16) * K1 + kb);

        acc0 = __builtin_amdgcn_wmma_f32_16x16x32_bf16(false, a, false, b0,
                                                       (short)0, acc0, false, false);
        acc1 = __builtin_amdgcn_wmma_f32_16x16x32_bf16(false, a, false, b1,
                                                       (short)0, acc1, false, false);
    }

    // ---- ReLU + restage through LDS (C layout -> A layout) ----
#pragma unroll
    for (int r = 0; r < 8; ++r) {
        float t0 = acc0[r] > 0.f ? acc0[r] : 0.f;
        float t1 = acc1[r] > 0.f ? acc1[r] : 0.f;
        sT[r + 8 * hi][mloc]      = t0;
        sT[r + 8 * hi][16 + mloc] = t1;
    }
    __syncthreads();

    // ---- GEMM 2: 16x32 @ 32x32, single K=32 step ----
    float bb0 = bB[mloc], bb1 = bB[16 + mloc];
    v8f d0, d1;
#pragma unroll
    for (int i = 0; i < 8; ++i) { d0[i] = bb0; d1[i] = bb1; }

    v16bf a2;
    {
        const int ka = hi * 16;
#pragma unroll
        for (int j = 0; j < 8; ++j) {
            a2[j]     = (__bf16)sT[mloc][ka + j];
            a2[j + 8] = (__bf16)sT[mloc][ka + 8 + j];
        }
    }
    const int kb2 = hi * 16;
    v16bf c0 = *reinterpret_cast<const v16bf*>(wBT + (size_t)mloc * 32 + kb2);
    v16bf c1 = *reinterpret_cast<const v16bf*>(wBT + (size_t)(mloc + 16) * 32 + kb2);

    d0 = __builtin_amdgcn_wmma_f32_16x16x32_bf16(false, a2, false, c0,
                                                 (short)0, d0, false, false);
    d1 = __builtin_amdgcn_wmma_f32_16x16x32_bf16(false, a2, false, c1,
                                                 (short)0, d1, false, false);

    // ---- store out (coalesced 64B row segments) ----
#pragma unroll
    for (int r = 0; r < 8; ++r) {
        int row = tile * 16 + r + 8 * hi;
        if (row < M) {
            out[(size_t)row * 32 + mloc]      = d0[r];
            out[(size_t)row * 32 + 16 + mloc] = d1[r];
        }
    }
}

// ---------------------------------------------------------------------------
// final: out[m] = agg[m,:] . w3 + b3      (w3 is 32x1)
// ---------------------------------------------------------------------------
__global__ void final_linear(const float* __restrict__ agg,
                             const float* __restrict__ w3,
                             const float* __restrict__ b3,
                             float* __restrict__ out, int M) {
    int m = blockIdx.x * blockDim.x + threadIdx.x;
    if (m >= M) return;
    const float* row = agg + (size_t)m * 32;
    float s = b3[0];
#pragma unroll
    for (int c = 0; c < 32; ++c) s += row[c] * w3[c];
    out[m] = s;
}

// ---------------------------------------------------------------------------
extern "C" void kernel_launch(void* const* d_in, const int* in_sizes, int n_in,
                              void* d_out, int out_size, void* d_ws, size_t ws_size,
                              hipStream_t stream) {
    const float* x   = (const float*)d_in[0];
    const int*   edg = (const int*)  d_in[1];
    const float* w1a = (const float*)d_in[2];
    const float* b1a = (const float*)d_in[3];
    const float* w1b = (const float*)d_in[4];
    const float* b1b = (const float*)d_in[5];
    const float* w2a = (const float*)d_in[6];
    const float* b2a = (const float*)d_in[7];
    const float* w2b = (const float*)d_in[8];
    const float* b2b = (const float*)d_in[9];
    const float* w3  = (const float*)d_in[10];
    const float* b3  = (const float*)d_in[11];
    float* out = (float*)d_out;

    const int M  = in_sizes[0] / 128;   // 100000 nodes
    const int nE = in_sizes[1] / 2;     // 1600000 edges

    // workspace carve (reuse: h2 aliases agg1, agg3 aliases h1)
    float* agg1 = (float*)d_ws;                    // M*128 f32
    float* h1   = agg1 + (size_t)M * 128;          // M*32  f32
    float* agg2 = h1   + (size_t)M * 32;           // M*32  f32
    float* h2   = agg1;
    float* agg3 = h1;
    // packed bf16 weights (32B-aligned: byte offset M*192*4 is mult. of 32)
    __bf16* w1aT = (__bf16*)(agg2 + (size_t)M * 32);   // 32*128
    __bf16* w1bT = w1aT + 32 * 128;                    // 32*32
    __bf16* w2aT = w1bT + 32 * 32;
    __bf16* w2bT = w2aT + 32 * 32;

    const int mlpBlocks  = (M + 15) / 16;
    const long long sthr = (long long)nE * 32;
    const int sBlocks    = (int)((sthr + 255) / 256);

    // ---- pack weights (tiny, once per launch) ----
    pack_w<<<(32 * 128 + 255) / 256, 256, 0, stream>>>(w1a, w1aT, 128);
    pack_w<<<(32 * 32 + 255) / 256, 256, 0, stream>>>(w1b, w1bT, 32);
    pack_w<<<(32 * 32 + 255) / 256, 256, 0, stream>>>(w2a, w2aT, 32);
    pack_w<<<(32 * 32 + 255) / 256, 256, 0, stream>>>(w2b, w2bT, 32);

    // ---- layer 1: agg = x + scatter(x); h1 = MLP128(agg) ----
    hipMemcpyAsync(agg1, x, (size_t)M * 128 * sizeof(float),
                   hipMemcpyDeviceToDevice, stream);
    scatter_add<<<sBlocks, 256, 0, stream>>>(x, agg1, edg, nE, 128);
    mlp_kernel<128><<<mlpBlocks, 32, 0, stream>>>(agg1, w1aT, b1a, w1bT, b1b, h1, M);

    // ---- layer 2 ----
    hipMemcpyAsync(agg2, h1, (size_t)M * 32 * sizeof(float),
                   hipMemcpyDeviceToDevice, stream);
    scatter_add<<<sBlocks, 256, 0, stream>>>(h1, agg2, edg, nE, 32);
    mlp_kernel<32><<<mlpBlocks, 32, 0, stream>>>(agg2, w2aT, b2a, w2bT, b2b, h2, M);

    // ---- layer 3 ----
    hipMemcpyAsync(agg3, h2, (size_t)M * 32 * sizeof(float),
                   hipMemcpyDeviceToDevice, stream);
    scatter_add<<<sBlocks, 256, 0, stream>>>(h2, agg3, edg, nE, 32);
    final_linear<<<(M + 255) / 256, 256, 0, stream>>>(agg3, w3, b3, out, M);
}
